// SelfContactOptiAnimAssignLoss_25340307046954
// MI455X (gfx1250) — compile-verified, matrix operand order checked by reading, not run on hardware
//
#include <hip/hip_runtime.h>
#include <math.h>

// ---------------- problem constants ----------------
#define NVV       10475
#define NDSN      2048
#define NHANDN    1554
#define NCONT     300
#define HA        (NHANDN / 2)          // 777
#define M_TOT     (NHANDN + NDSN)       // 3602 (hand queries first, then ds)
#define QTILES    ((M_TOT + 15) / 16)   // 226
#define M_PAD     (QTILES * 16)         // 3616
#define COLTILES  ((NVV + 15) / 16)     // 655
#define WAVES_PB  8
#define ITERS     8
#define TILES_PB  (WAVES_PB * ITERS)    // 64 column tiles per block
#define GYCHUNKS  ((COLTILES + TILES_PB - 1) / TILES_PB) // 11

#define GEO_THRESH 0.3f
#define EXT_THRESH 0.02f
#define A1c 0.04f
#define A2c 0.04f
#define B1c 0.07f
#define B2c 0.06f
#define C1c 0.01f
#define C2c 0.01f
#define D1c 0.023f
#define D2c 0.02f

typedef __attribute__((ext_vector_type(16))) _Float16 v16h;
typedef __attribute__((ext_vector_type(8)))  float    v8f;

__device__ inline float posinf() { return __uint_as_float(0x7f800000u); }

// ---------------- kernel 1: vertex prep ----------------
// fp16-round coords (reference: astype(fp16)), precompute |v|^2 (f32 of rounded
// values), zero the inside-flag array.
__global__ void prep_vertices_k(const float* __restrict__ vertices,
                                _Float16* __restrict__ vhx, _Float16* __restrict__ vhy,
                                _Float16* __restrict__ vhz, float* __restrict__ vn2,
                                int* __restrict__ inside) {
  int i = blockIdx.x * blockDim.x + threadIdx.x;
  if (i >= NVV) return;
  _Float16 hx = (_Float16)vertices[3 * i + 0];   // RTNE, matches astype(fp16)
  _Float16 hy = (_Float16)vertices[3 * i + 1];
  _Float16 hz = (_Float16)vertices[3 * i + 2];
  vhx[i] = hx; vhy[i] = hy; vhz[i] = hz;
  float fx = (float)hx, fy = (float)hy, fz = (float)hz;
  vn2[i] = fx * fx + fy * fy + fz * fz;
  inside[i] = 0;
}

// ---------------- kernel 2: query prep ----------------
// Gather [hand_idx ; ds] queries, store (-2*q) fp16 A-fragment data, |q|^2,
// geodist row index; init per-query min(d^2) accumulator to +inf (every call).
__global__ void prep_queries_k(const int* __restrict__ hand_idx, const int* __restrict__ ds,
                               const _Float16* __restrict__ vhx, const _Float16* __restrict__ vhy,
                               const _Float16* __restrict__ vhz, const float* __restrict__ vn2,
                               _Float16* __restrict__ qhx, _Float16* __restrict__ qhy,
                               _Float16* __restrict__ qhz, float* __restrict__ qn2,
                               int* __restrict__ qrow, unsigned int* __restrict__ minD2u) {
  int m = blockIdx.x * blockDim.x + threadIdx.x;
  if (m >= M_PAD) return;
  int qidx = 0;
  if (m < NHANDN)      qidx = hand_idx[m];
  else if (m < M_TOT)  qidx = ds[m - NHANDN];
  float fx = (float)vhx[qidx], fy = (float)vhy[qidx], fz = (float)vhz[qidx];
  qhx[m] = (_Float16)(-2.0f * fx);   // exact power-of-2 scale in fp16
  qhy[m] = (_Float16)(-2.0f * fy);
  qhz[m] = (_Float16)(-2.0f * fz);
  qn2[m] = vn2[qidx];
  qrow[m] = qidx;
  minD2u[m] = 0x7f800000u;           // +inf
}

// ---------------- kernel 3: WMMA masked min-distance ----------------
// Block = 8 waves; each block owns one 16-query tile, each wave streams 8
// 16-column tiles. d^2 tile = WMMA(-2q, v) + (|q|^2 + |v|^2) in C.
__global__ __launch_bounds__(256) void minpair_k(
    const _Float16* __restrict__ qhx, const _Float16* __restrict__ qhy,
    const _Float16* __restrict__ qhz, const float* __restrict__ qn2,
    const int* __restrict__ qrow,
    const _Float16* __restrict__ vhx, const _Float16* __restrict__ vhy,
    const _Float16* __restrict__ vhz, const float* __restrict__ vn2,
    const float* __restrict__ geodist,
    unsigned int* __restrict__ minD2u) {
  __shared__ int      s_row[16];
  __shared__ float    s_qn2[16];
  __shared__ _Float16 s_qx[16], s_qy[16], s_qz[16];

  const int tileBase = blockIdx.x * 16;
  const int t = threadIdx.x;
  if (t < 16) {
    int m = tileBase + t;
    s_row[t] = qrow[m];
    s_qn2[t] = qn2[m];
    s_qx[t] = qhx[m]; s_qy[t] = qhy[m]; s_qz[t] = qhz[m];
  }
  __syncthreads();

  const int wave = t >> 5;
  const int lane = t & 31;
  const int lo = lane & 15;        // N within tile (C layout)
  const int hi = lane >> 4;        // M offset selector (C layout: M = r + 8*hi)

  // A fragment: 16x32 f16, only K=0..2 nonzero. Lanes 0-15 hold M=lane, K=0..7
  // in elements 0..7; lanes 16-31 hold K=8..15 -> all zero.
  v16h a;
  #pragma unroll
  for (int i = 0; i < 16; ++i) a[i] = (_Float16)0;
  if (lane < 16) { a[0] = s_qx[lane]; a[1] = s_qy[lane]; a[2] = s_qz[lane]; }

  float macc[8];
  #pragma unroll
  for (int r = 0; r < 8; ++r) macc[r] = posinf();

  const int tile0 = (blockIdx.y * WAVES_PB + wave) * ITERS;
  for (int it = 0; it < ITERS; ++it) {
    int jbase = (tile0 + it) * 16;
    if (jbase >= NVV) break;       // wave-uniform

    // B fragment: 32x16 f16 (KxN), only K=0..2 nonzero; lanes 0-15 hold N=lane.
    v16h b;
    #pragma unroll
    for (int i = 0; i < 16; ++i) b[i] = (_Float16)0;
    if (lane < 16) {
      int jv = jbase + lane; int jc = jv < NVV ? jv : NVV - 1;
      b[0] = vhx[jc]; b[1] = vhy[jc]; b[2] = vhz[jc];
    }

    int jcol = jbase + lo;
    int jcc  = jcol < NVV ? jcol : NVV - 1;
    float vv = vn2[jcc];

    v8f c;
    #pragma unroll
    for (int r = 0; r < 8; ++r) c[r] = s_qn2[r + 8 * hi] + vv;

    // d^2 tile: |q|^2 + |v|^2 - 2 q.v  (EXEC all-1s here)
    c = __builtin_amdgcn_wmma_f32_16x16x32_f16(false, a, false, b,
                                               (short)0, c, false, false);

    bool inb = (jcol < NVV);
    #pragma unroll
    for (int r = 0; r < 8; ++r) {
      size_t row = (size_t)s_row[r + 8 * hi];
      float g = geodist[row * (size_t)NVV + (size_t)jcc];   // coalesced 64B/halfwave
      float d2 = fmaxf(c[r], 0.0f);                         // guard cancellation
      macc[r] = fminf(macc[r], (inb && g >= GEO_THRESH) ? d2 : posinf());
    }
  }

  // min over the 16 N-lanes of each half-wave, then global atomic min
  #pragma unroll
  for (int r = 0; r < 8; ++r) {
    float v = macc[r];
    v = fminf(v, __shfl_xor(v, 1, 32));
    v = fminf(v, __shfl_xor(v, 2, 32));
    v = fminf(v, __shfl_xor(v, 4, 32));
    v = fminf(v, __shfl_xor(v, 8, 32));
    if (lo == 0)
      atomicMin(&minD2u[tileBase + r + 8 * hi], __float_as_uint(v)); // v >= 0 or +inf
  }
}

// ---------------- kernel 4: gdi at ds rows ----------------
__global__ void gdi_k(const float* __restrict__ geodist, const int* __restrict__ ds,
                      const int* __restrict__ cidx, float* __restrict__ gdi_ds) {
  int gwave = (blockIdx.x * blockDim.x + threadIdx.x) >> 5;
  int lane = threadIdx.x & 31;
  if (gwave >= NDSN) return;
  size_t base = (size_t)ds[gwave] * (size_t)NVV;
  float mn = posinf();
  for (int k = lane; k < NCONT; k += 32) mn = fminf(mn, geodist[base + cidx[k]]);
  mn = fminf(mn, __shfl_xor(mn, 1, 32));
  mn = fminf(mn, __shfl_xor(mn, 2, 32));
  mn = fminf(mn, __shfl_xor(mn, 4, 32));
  mn = fminf(mn, __shfl_xor(mn, 8, 32));
  mn = fminf(mn, __shfl_xor(mn, 16, 32));
  if (lane == 0) gdi_ds[gwave] = mn;
}

// ---------------- kernel 5: inside scatter ----------------
__global__ void scatter_inside_k(const unsigned int* __restrict__ minD2u,
                                 const int* __restrict__ ds, int* __restrict__ inside) {
  int i = blockIdx.x * blockDim.x + threadIdx.x;
  if (i >= NDSN) return;
  float d2 = fmaxf(__uint_as_float(minD2u[NHANDN + i]), 0.0f);
  float d = sqrtf(d2 + 1e-12f);
  if (!(d > EXT_THRESH)) atomicOr(&inside[ds[i]], 1);   // ~exterior
}

// ---------------- kernel 6: finalize ----------------
__device__ float block_sum(float v, float* sbuf) {
  int t = threadIdx.x;
  sbuf[t] = v; __syncthreads();
  for (int s = 128; s > 0; s >>= 1) {
    if (t < s) sbuf[t] += sbuf[t + s];
    __syncthreads();
  }
  float r = sbuf[0];
  __syncthreads();
  return r;
}

__global__ __launch_bounds__(256) void finalize_k(
    const unsigned int* __restrict__ minD2u, const float* __restrict__ gdi_ds,
    const int* __restrict__ inside, const int* __restrict__ ds,
    const int* __restrict__ hand_idx, const float* __restrict__ hand_weights,
    float* __restrict__ out) {
  __shared__ float sbuf[256];
  int t = threadIdx.x;

  float cs = 0, cc = 0, ins_s = 0, ins_c = 0;
  for (int i = t; i < NDSN; i += 256) {
    float d2 = fmaxf(__uint_as_float(minD2u[NHANDN + i]), 0.0f);
    float d = sqrtf(d2 + 1e-12f);
    bool in = inside[ds[i]] != 0;
    float w = 1.0f / (5.0f * gdi_ds[i] + 1.0f);
    if (!in) { cs += A1c * w * tanhf(d / A2c); cc += 1.0f; }
    else     { ins_s += B1c * tanhf(d / B2c);  ins_c += 1.0f; }
  }

  float los = 0, loc = 0, ros = 0, roc = 0, lis = 0, lic = 0, ris = 0, ric = 0;
  for (int i = t; i < NHANDN; i += 256) {
    float d2 = fmaxf(__uint_as_float(minD2u[i]), 0.0f);
    float d = sqrtf(d2 + 1e-12f);
    bool in = inside[hand_idx[i]] != 0;
    bool left = i < HA;
    float w = 1.0f - 0.1f * hand_weights[i];
    float o = C1c * tanhf(d / C2c);
    float ii = D1c * tanhf(d / D2c);
    if (!in) {
      if (left) { los += w * o; loc += 1.0f; } else { ros += w * o; roc += 1.0f; }
    } else {
      if (left) { lis += ii; lic += 1.0f; } else { ris += ii; ric += 1.0f; }
    }
  }

  cs = block_sum(cs, sbuf); cc = block_sum(cc, sbuf);
  ins_s = block_sum(ins_s, sbuf); ins_c = block_sum(ins_c, sbuf);
  los = block_sum(los, sbuf); loc = block_sum(loc, sbuf);
  ros = block_sum(ros, sbuf); roc = block_sum(roc, sbuf);
  lis = block_sum(lis, sbuf); lic = block_sum(lic, sbuf);
  ris = block_sum(ris, sbuf); ric = block_sum(ric, sbuf);

  if (t == 0) {
    float contact = 0.5f * (cs / fmaxf(cc, 1.0f));
    float insideL = 0.5f * (ins_s / fmaxf(ins_c, 1.0f));
    float hand = los / fmaxf(loc, 1.0f) + ros / fmaxf(roc, 1.0f) +
                 lis / fmaxf(lic, 1.0f) + ris / fmaxf(ric, 1.0f);
    out[0] = contact + insideL + hand;
  }
}

// ---------------- launcher ----------------
extern "C" void kernel_launch(void* const* d_in, const int* in_sizes, int n_in,
                              void* d_out, int out_size, void* d_ws, size_t ws_size,
                              hipStream_t stream) {
  const float* vertices     = (const float*)d_in[0];  // [1,NV,3] f32
  const float* geodist      = (const float*)d_in[1];  // [NV,NV] f32
  const float* hand_weights = (const float*)d_in[2];  // [NHAND] f32
  const int*   ds           = (const int*)d_in[3];    // [NDS] i32
  const int*   hand_idx     = (const int*)d_in[4];    // [NHAND] i32
  const int*   cidx         = (const int*)d_in[5];    // [NCONTACT] i32
  float*       out          = (float*)d_out;

  // workspace carve (256B aligned slices)
  char* p = (char*)d_ws;
  auto carve = [&](size_t bytes) -> char* {
    char* r = p;
    p += (bytes + 255) & ~(size_t)255;
    return r;
  };
  _Float16* vhx = (_Float16*)carve(NVV * sizeof(_Float16));
  _Float16* vhy = (_Float16*)carve(NVV * sizeof(_Float16));
  _Float16* vhz = (_Float16*)carve(NVV * sizeof(_Float16));
  float*    vn2 = (float*)carve(NVV * sizeof(float));
  int*   inside = (int*)carve(NVV * sizeof(int));
  _Float16* qhx = (_Float16*)carve(M_PAD * sizeof(_Float16));
  _Float16* qhy = (_Float16*)carve(M_PAD * sizeof(_Float16));
  _Float16* qhz = (_Float16*)carve(M_PAD * sizeof(_Float16));
  float*    qn2 = (float*)carve(M_PAD * sizeof(float));
  int*     qrow = (int*)carve(M_PAD * sizeof(int));
  unsigned int* minD2u = (unsigned int*)carve(M_PAD * sizeof(unsigned int));
  float* gdi_ds = (float*)carve(NDSN * sizeof(float));

  prep_vertices_k<<<(NVV + 255) / 256, 256, 0, stream>>>(vertices, vhx, vhy, vhz, vn2, inside);
  prep_queries_k<<<(M_PAD + 255) / 256, 256, 0, stream>>>(hand_idx, ds, vhx, vhy, vhz, vn2,
                                                          qhx, qhy, qhz, qn2, qrow, minD2u);
  dim3 grid(QTILES, GYCHUNKS);
  minpair_k<<<grid, 256, 0, stream>>>(qhx, qhy, qhz, qn2, qrow,
                                      vhx, vhy, vhz, vn2, geodist, minD2u);
  gdi_k<<<(NDSN * 32 + 255) / 256, 256, 0, stream>>>(geodist, ds, cidx, gdi_ds);
  scatter_inside_k<<<(NDSN + 255) / 256, 256, 0, stream>>>(minD2u, ds, inside);
  finalize_k<<<1, 256, 0, stream>>>(minD2u, gdi_ds, inside, ds, hand_idx, hand_weights, out);
}
